// SwiGLU_36773509989064
// MI455X (gfx1250) — compile-verified, matrix operand order checked by reading, not run on hardware
//
#include <hip/hip_runtime.h>

typedef __attribute__((ext_vector_type(16))) __bf16 v16bf;
typedef __attribute__((ext_vector_type(8)))  __bf16 v8bf;
typedef __attribute__((ext_vector_type(4)))  __bf16 v4bf;
typedef __attribute__((ext_vector_type(8)))  float  v8f;

#define EPSQ 1e-8f

// ---------------- CDNA5 async copy helpers ----------------

// 16-byte global -> LDS async copy (ASYNCcnt-tracked, bypasses VGPR staging).
__device__ __forceinline__ void async_cp16(unsigned lds_byte_off, const __bf16* gptr) {
    asm volatile("global_load_async_to_lds_b128 %0, %1, off"
                 :
                 : "v"(lds_byte_off), "v"((unsigned long long)(uintptr_t)gptr)
                 : "memory");
}

__device__ __forceinline__ void wait_async0() {
    asm volatile("s_wait_asynccnt 0x0" ::: "memory");
}

// LDS byte offset of a __shared__ pointer (generic addr low 32 bits == LDS offset)
__device__ __forceinline__ unsigned lds_off(const void* p) {
    return (unsigned)(uintptr_t)p;
}

// ---------------- gamma = mean(|W|) helpers ----------------

__global__ void init_gamma_kernel(float* g) {
    if (threadIdx.x < 3) g[threadIdx.x] = 0.0f;
}

__global__ __launch_bounds__(256) void abs_sum_kernel(const float* __restrict__ w,
                                                      long long n,
                                                      float* __restrict__ out) {
    __shared__ float red[256];
    const long long nv = n >> 2;
    float s = 0.0f;
    for (long long i = (long long)blockIdx.x * 256 + threadIdx.x; i < nv;
         i += (long long)gridDim.x * 256) {
        float4 v = ((const float4*)w)[i];
        s += fabsf(v.x) + fabsf(v.y) + fabsf(v.z) + fabsf(v.w);
    }
    red[threadIdx.x] = s;
    __syncthreads();
    for (int off = 128; off > 0; off >>= 1) {
        if ((int)threadIdx.x < off) red[threadIdx.x] += red[threadIdx.x + off];
        __syncthreads();
    }
    if (threadIdx.x == 0) atomicAdd(out, red[0]);
}

// W_hat = clamp(round(W/(gamma+eps)), -1, 1)  -> ternary, exact in bf16
__global__ __launch_bounds__(256) void quantize_kernel(const float* __restrict__ w,
                                                       __bf16* __restrict__ q,
                                                       long long n,
                                                       const float* __restrict__ gsum) {
    const float gamma = (*gsum) / (float)n;
    const float inv = 1.0f / (gamma + EPSQ);
    const long long nv = n >> 2;
    for (long long i = (long long)blockIdx.x * 256 + threadIdx.x; i < nv;
         i += (long long)gridDim.x * 256) {
        float4 v = ((const float4*)w)[i];
        v4bf o;
        float t;
        t = fminf(1.0f, fmaxf(-1.0f, rintf(v.x * inv))); o[0] = (__bf16)t;
        t = fminf(1.0f, fmaxf(-1.0f, rintf(v.y * inv))); o[1] = (__bf16)t;
        t = fminf(1.0f, fmaxf(-1.0f, rintf(v.z * inv))); o[2] = (__bf16)t;
        t = fminf(1.0f, fmaxf(-1.0f, rintf(v.w * inv))); o[3] = (__bf16)t;
        ((v4bf*)q)[i] = o;
    }
}

__global__ __launch_bounds__(256) void cvt_bf16_kernel(const float* __restrict__ x,
                                                       __bf16* __restrict__ xb,
                                                       long long n) {
    const long long nv = n >> 2;
    for (long long i = (long long)blockIdx.x * 256 + threadIdx.x; i < nv;
         i += (long long)gridDim.x * 256) {
        float4 v = ((const float4*)x)[i];
        v4bf o;
        o[0] = (__bf16)v.x; o[1] = (__bf16)v.y; o[2] = (__bf16)v.z; o[3] = (__bf16)v.w;
        ((v4bf*)xb)[i] = o;
    }
}

// ---------------- WMMA GEMM tiles ----------------
// Block tile: 128(M) x 64(N), K-step 32, double-buffered LDS filled by
// async global->LDS copies.  256 threads = 8 waves (wave32), wave grid 4x2,
// each wave owns a 32x32 tile = 2x2 WMMA 16x16x32 bf16 tiles.

#define BM 128
#define BN 64
#define BK 32
#define LDPAD 40   // halves per LDS row (32 + 8 pad); 80B row stride, 16B aligned

// 16-bit A-matrix 16x32 fragment layout (ISA 7.12.2):
// lanes 0-15: M=lane, K-chunks {0..7, 16..23}; lanes 16-31: M=lane-16, K {8..15, 24..31}
__device__ __forceinline__ v16bf load_frag(const __bf16* base, int row0, int lane) {
    int r  = row0 + (lane & 15);
    int kc = (lane >> 4) << 3;   // 0 or 8
    const __bf16* p = base + r * LDPAD + kc;
    v8bf lo = *(const v8bf*)(p);
    v8bf hi = *(const v8bf*)(p + 16);
    return __builtin_shufflevector(lo, hi, 0, 1, 2, 3, 4, 5, 6, 7,
                                           8, 9, 10, 11, 12, 13, 14, 15);
}

// Issue async loads of one 128x32 A tile into LDS buffer at aOff.
__device__ __forceinline__ void stage_a(const __bf16* __restrict__ A, int m0, int K,
                                        int k0, unsigned aOff, int tid) {
    #pragma unroll
    for (int it = 0; it < 2; ++it) {
        int idx = tid + it * 256;
        int r = idx >> 2;
        int c = (idx & 3) << 3;
        async_cp16(aOff + (unsigned)(r * LDPAD + c) * 2,
                   A + (size_t)(m0 + r) * K + k0 + c);
    }
}

// Issue async loads of one 64x32 B tile into LDS buffer at bOff.
__device__ __forceinline__ void stage_b(const __bf16* __restrict__ B, int n0, int K,
                                        int k0, unsigned bOff, int tid) {
    int r = tid >> 2;
    int c = (tid & 3) << 3;
    async_cp16(bOff + (unsigned)(r * LDPAD + c) * 2,
               B + (size_t)(n0 + r) * K + k0 + c);
}

// hidden = sigmoid(x @ w1q^T) * (x @ w2q^T), output bf16 [M,N]
__global__ __launch_bounds__(256) void gemm_gated_kernel(
    const __bf16* __restrict__ A,    // [M,K]
    const __bf16* __restrict__ B1,   // [N,K]
    const __bf16* __restrict__ B2,   // [N,K]
    __bf16* __restrict__ Hout,       // [M,N]
    int M, int N, int K) {
    __shared__ __bf16 As[2][BM * LDPAD];
    __shared__ __bf16 B1s[2][BN * LDPAD];
    __shared__ __bf16 B2s[2][BN * LDPAD];

    const int tid   = threadIdx.x;
    const int lane  = tid & 31;
    const int wave  = tid >> 5;
    const int waveM = wave & 3;   // 0..3
    const int waveN = wave >> 2;  // 0..1
    const int m0 = blockIdx.y * BM;
    const int n0 = blockIdx.x * BN;

    const v8f vzero = {0.f, 0.f, 0.f, 0.f, 0.f, 0.f, 0.f, 0.f};
    v8f acc1[2][2], acc2[2][2];
    for (int i = 0; i < 2; ++i)
        for (int j = 0; j < 2; ++j) { acc1[i][j] = vzero; acc2[i][j] = vzero; }

    const int nk = K / BK;
    // prologue: fill buffer 0
    stage_a(A, m0, K, 0, lds_off(&As[0][0]), tid);
    stage_b(B1, n0, K, 0, lds_off(&B1s[0][0]), tid);
    stage_b(B2, n0, K, 0, lds_off(&B2s[0][0]), tid);

    for (int i = 0; i < nk; ++i) {
        const int cur = i & 1;
        const int nxt = cur ^ 1;
        wait_async0();      // this wave's pending loads (current buffer) done
        __syncthreads();    // => all waves' loads done; prev readers finished
        if (i + 1 < nk) {
            int k1 = (i + 1) * BK;
            stage_a(A, m0, K, k1, lds_off(&As[nxt][0]), tid);
            stage_b(B1, n0, K, k1, lds_off(&B1s[nxt][0]), tid);
            stage_b(B2, n0, K, k1, lds_off(&B2s[nxt][0]), tid);
        }
        v16bf af[2], b1f[2], b2f[2];
        #pragma unroll
        for (int a = 0; a < 2; ++a) af[a]  = load_frag(&As[cur][0],  waveM * 32 + a * 16, lane);
        #pragma unroll
        for (int b = 0; b < 2; ++b) { b1f[b] = load_frag(&B1s[cur][0], waveN * 32 + b * 16, lane);
                                      b2f[b] = load_frag(&B2s[cur][0], waveN * 32 + b * 16, lane); }
        #pragma unroll
        for (int a = 0; a < 2; ++a)
            #pragma unroll
            for (int b = 0; b < 2; ++b) {
                acc1[a][b] = __builtin_amdgcn_wmma_f32_16x16x32_bf16(
                    false, af[a], false, b1f[b], (short)0, acc1[a][b], false, false);
                acc2[a][b] = __builtin_amdgcn_wmma_f32_16x16x32_bf16(
                    false, af[a], false, b2f[b], (short)0, acc2[a][b], false, false);
            }
    }

    // Epilogue: C layout — VGPR e, lanes 0-15: M=e,N=lane; lanes 16-31: M=8+e,N=lane-16
    const int ncol  = lane & 15;
    const int mbase = (lane >> 4) << 3;
    #pragma unroll
    for (int a = 0; a < 2; ++a)
        #pragma unroll
        for (int b = 0; b < 2; ++b) {
            int gm = m0 + waveM * 32 + a * 16 + mbase;
            int gn = n0 + waveN * 32 + b * 16 + ncol;
            #pragma unroll
            for (int e = 0; e < 8; ++e) {
                float a1 = acc1[a][b][e];
                float a2 = acc2[a][b][e];
                // fast sigmoid: raw v_rcp_f32, no IEEE div fixup
                float hv = a2 * __builtin_amdgcn_rcpf(1.0f + __expf(-a1));
                Hout[(size_t)(gm + e) * N + gn] = (__bf16)hv;
            }
        }
}

// out_f32 = hidden @ w3q^T
__global__ __launch_bounds__(256) void gemm_out_kernel(
    const __bf16* __restrict__ A,   // [M,K]
    const __bf16* __restrict__ B,   // [N,K]
    float* __restrict__ Out,        // [M,N]
    int M, int N, int K) {
    __shared__ __bf16 As[2][BM * LDPAD];
    __shared__ __bf16 Bs[2][BN * LDPAD];

    const int tid   = threadIdx.x;
    const int lane  = tid & 31;
    const int wave  = tid >> 5;
    const int waveM = wave & 3;
    const int waveN = wave >> 2;
    const int m0 = blockIdx.y * BM;
    const int n0 = blockIdx.x * BN;

    const v8f vzero = {0.f, 0.f, 0.f, 0.f, 0.f, 0.f, 0.f, 0.f};
    v8f acc[2][2];
    for (int i = 0; i < 2; ++i)
        for (int j = 0; j < 2; ++j) acc[i][j] = vzero;

    const int nk = K / BK;
    stage_a(A, m0, K, 0, lds_off(&As[0][0]), tid);
    stage_b(B, n0, K, 0, lds_off(&Bs[0][0]), tid);

    for (int i = 0; i < nk; ++i) {
        const int cur = i & 1;
        const int nxt = cur ^ 1;
        wait_async0();
        __syncthreads();
        if (i + 1 < nk) {
            int k1 = (i + 1) * BK;
            stage_a(A, m0, K, k1, lds_off(&As[nxt][0]), tid);
            stage_b(B, n0, K, k1, lds_off(&Bs[nxt][0]), tid);
        }
        v16bf af[2], bf[2];
        #pragma unroll
        for (int a = 0; a < 2; ++a) af[a] = load_frag(&As[cur][0], waveM * 32 + a * 16, lane);
        #pragma unroll
        for (int b = 0; b < 2; ++b) bf[b] = load_frag(&Bs[cur][0], waveN * 32 + b * 16, lane);
        #pragma unroll
        for (int a = 0; a < 2; ++a)
            #pragma unroll
            for (int b = 0; b < 2; ++b)
                acc[a][b] = __builtin_amdgcn_wmma_f32_16x16x32_bf16(
                    false, af[a], false, bf[b], (short)0, acc[a][b], false, false);
    }

    const int ncol  = lane & 15;
    const int mbase = (lane >> 4) << 3;
    #pragma unroll
    for (int a = 0; a < 2; ++a)
        #pragma unroll
        for (int b = 0; b < 2; ++b) {
            int gm = m0 + waveM * 32 + a * 16 + mbase;
            int gn = n0 + waveN * 32 + b * 16 + ncol;
            #pragma unroll
            for (int e = 0; e < 8; ++e)
                Out[(size_t)(gm + e) * N + gn] = acc[a][b][e];
        }
}

// ---------------- launch ----------------

extern "C" void kernel_launch(void* const* d_in, const int* in_sizes, int n_in,
                              void* d_out, int out_size, void* d_ws, size_t ws_size,
                              hipStream_t stream) {
    const float* x  = (const float*)d_in[0];   // [4,2048,2048]
    const float* w1 = (const float*)d_in[1];   // [5632,2048]
    const float* w2 = (const float*)d_in[2];   // [5632,2048]
    const float* w3 = (const float*)d_in[3];   // [2048,5632]
    float* out = (float*)d_out;                // [4,2048,2048]

    const long long Mr = 8192, Dd = 2048, Hh = 5632;
    const long long HD = Hh * Dd;   // weight elements
    const long long MD = Mr * Dd;   // x / out elements
    const long long MH = Mr * Hh;   // hidden elements

    char* base = (char*)d_ws;
    float*  gsum = (float*)base;                           // 3 floats (256B reserved)
    __bf16* hid  = (__bf16*)(base + 256);                  // M*H bf16
    __bf16* xb   = (__bf16*)(base + 256 + (size_t)MH * 2); // M*D bf16
    __bf16* w1q  = xb + MD;                                // H*D bf16
    __bf16* w2q  = w1q + HD;                               // H*D bf16
    __bf16* w3q  = xb;  // aliases xb: written only after gemm_gated consumed xb

    init_gamma_kernel<<<1, 32, 0, stream>>>(gsum);
    abs_sum_kernel<<<1024, 256, 0, stream>>>(w1, HD, gsum + 0);
    abs_sum_kernel<<<1024, 256, 0, stream>>>(w2, HD, gsum + 1);
    abs_sum_kernel<<<1024, 256, 0, stream>>>(w3, HD, gsum + 2);

    quantize_kernel<<<2048, 256, 0, stream>>>(w1, w1q, HD, gsum + 0);
    quantize_kernel<<<2048, 256, 0, stream>>>(w2, w2q, HD, gsum + 1);
    cvt_bf16_kernel<<<2048, 256, 0, stream>>>(x, xb, MD);

    gemm_gated_kernel<<<dim3(Hh / BN, Mr / BM), 256, 0, stream>>>(
        xb, w1q, w2q, hid, (int)Mr, (int)Hh, (int)Dd);

    quantize_kernel<<<2048, 256, 0, stream>>>(w3, w3q, HD, gsum + 2);

    gemm_out_kernel<<<dim3(Dd / BN, Mr / BM), 256, 0, stream>>>(
        hid, w3q, out, (int)Mr, (int)Dd, (int)Hh);
}
